// WindingGraphNet_37177236914580
// MI455X (gfx1250) — matrix operand order repaired and checked by reading, not compiled, not run on hardware
//
#include <hip/hip_runtime.h>
#include <hip/hip_bf16.h>

typedef __attribute__((ext_vector_type(16))) __bf16 v16bf;
typedef __attribute__((ext_vector_type(8)))  __bf16 v8bf;
typedef __attribute__((ext_vector_type(8)))  float  v8f;

static constexpr long gN = 20000;
static constexpr long gE = 320000;

// ---------------- param indices (setup_inputs insertion order) ----------------
enum {
  IN_X = 0, IN_EATTR, IN_U, IN_HX, IN_HE, IN_HU, IN_EPSE, IN_EPSN, IN_EIDX, IN_BATCH,
  P_EE_W1, P_EE_B1, P_EE_W2, P_EE_B2,
  P_EN_W1, P_EN_B1, P_EN_W2, P_EN_B2,
  P_EG_W1, P_EG_B1, P_EG_W2, P_EG_B2,
  P_RE_WZ, P_RE_BZ, P_RE_WR, P_RE_BR, P_RE_WN, P_RE_BN,
  P_RN_WZ, P_RN_BZ, P_RN_WR, P_RN_BR, P_RN_WN, P_RN_BN,
  P_RG_WZ, P_RG_BZ, P_RG_WR, P_RG_BR, P_RG_WN, P_RG_BN,
  P_DE_W1, P_DE_B1, P_DE_W2, P_DE_B2,
  P_DN_W1, P_DN_B1, P_DN_W2, P_DN_B2,
  P_DG_W1, P_DG_B1, P_DG_W2, P_DG_B2
};

// ---------------- WMMA micro-kernel ----------------
// A: LDS, row-major 64 x lda (bf16).  Bt: transposed weights (Nout x ldb, bf16, global).
// Computes one 16x16 f32 tile at (mt, nt), K-loop of KK (multiple of 32).
template<int KK>
__device__ inline v8f wmma_loop(const __bf16* __restrict__ A, int lda,
                                const __bf16* __restrict__ Bt, int ldb,
                                int mt, int nt, int lane) {
  v8f acc = {};
  const __bf16* ap = A  + (long)(mt * 16 + (lane & 15)) * lda + ((lane >> 4) << 3);
  const __bf16* bp = Bt + (long)(nt * 16 + (lane & 15)) * ldb + ((lane >> 4) << 3);
#pragma unroll
  for (int k0 = 0; k0 < KK; k0 += 32) {
    v8bf a0 = *(const v8bf*)(ap + k0);
    v8bf a1 = *(const v8bf*)(ap + k0 + 16);
    v8bf b0 = *(const v8bf*)(bp + k0);
    v8bf b1 = *(const v8bf*)(bp + k0 + 16);
    v16bf a = __builtin_shufflevector(a0, a1, 0,1,2,3,4,5,6,7,8,9,10,11,12,13,14,15);
    v16bf b = __builtin_shufflevector(b0, b1, 0,1,2,3,4,5,6,7,8,9,10,11,12,13,14,15);
    acc = __builtin_amdgcn_wmma_f32_16x16x32_bf16(false, a, false, b, (short)0, acc, false, false);
  }
  return acc;
}

__device__ inline float sigmf(float x) { return 1.f / (1.f + expf(-x)); }

// ---------------- fused 2-layer MLP cores (HID = 128) ----------------
// GEMM1: A(64xK1) @ W1t^T -> relu -> H(64x128 bf16 in LDS)
// GEMM2 (64-wide out): H @ W2t^T + b2 -> global (row-guarded)
template<int K1>
__device__ void mlp_core64(const __bf16* A, __bf16* H,
                           const __bf16* W1t, const float* b1,
                           const __bf16* W2t, const float* b2,
                           float* gOut, long rowBase, long rows, int tid) {
  int w = tid >> 5, lane = tid & 31;
#pragma unroll
  for (int mt = 0; mt < 4; ++mt) {
    v8f acc = wmma_loop<K1>(A, K1, W1t, K1, mt, w, lane);
    int n = w * 16 + (lane & 15);
    int mbase = mt * 16 + ((lane >> 4) << 3);
    float bb = b1[n];
#pragma unroll
    for (int q = 0; q < 8; ++q) {
      float h = acc[q] + bb;
      H[(mbase + q) * 128 + n] = (__bf16)(h > 0.f ? h : 0.f);
    }
  }
  __syncthreads();
  int nt = w & 3, mt0 = (w >> 2) * 2;
#pragma unroll
  for (int mi = 0; mi < 2; ++mi) {
    int mt = mt0 + mi;
    v8f acc = wmma_loop<128>(H, 128, W2t, 128, mt, nt, lane);
    int n = nt * 16 + (lane & 15);
    int mbase = mt * 16 + ((lane >> 4) << 3);
    float bb = b2[n];
#pragma unroll
    for (int q = 0; q < 8; ++q) {
      long r = rowBase + mbase + q;
      if (r < rows) gOut[r * 64 + n] = acc[q] + bb;
    }
  }
}

// GEMM2 (4-wide out, padded to 16): stats tile -> LDS (64x16 f32)
template<int K1>
__device__ void mlp_core16(const __bf16* A, __bf16* H,
                           const __bf16* W1t, const float* b1,
                           const __bf16* W2t, const float* b2,
                           float* OutLds, int tid) {
  int w = tid >> 5, lane = tid & 31;
#pragma unroll
  for (int mt = 0; mt < 4; ++mt) {
    v8f acc = wmma_loop<K1>(A, K1, W1t, K1, mt, w, lane);
    int n = w * 16 + (lane & 15);
    int mbase = mt * 16 + ((lane >> 4) << 3);
    float bb = b1[n];
#pragma unroll
    for (int q = 0; q < 8; ++q) {
      float h = acc[q] + bb;
      H[(mbase + q) * 128 + n] = (__bf16)(h > 0.f ? h : 0.f);
    }
  }
  __syncthreads();
  if (w < 4) {
    v8f acc = wmma_loop<128>(H, 128, W2t, 128, w, 0, lane);
    int n = lane & 15;
    int mbase = w * 16 + ((lane >> 4) << 3);
    float bb = (n < 4) ? b2[n] : 0.f;
#pragma unroll
    for (int q = 0; q < 8; ++q) OutLds[(mbase + q) * 16 + n] = acc[q] + bb;
  }
  __syncthreads();
}

// ---------------- fused GRU core ----------------
// A(64xK) = cat(x, h); z/r from full K; n-gate = A[:, :DIN]@Wn[:DIN] + (r*h)@Wn[DIN:].
template<int K, int DIN>
__device__ void gru_core(const __bf16* A, __bf16* RH,
                         const __bf16* Wzt, const __bf16* Wrt, const __bf16* Wnt,
                         const float* bz, const float* br, const float* bn,
                         const float* hsrc, float* outp,
                         long rowBase, int rowsValid, int tid) {
  int w = tid >> 5, lane = tid & 31;
  v8f zs[2], ps[2];
#pragma unroll
  for (int t = 0; t < 2; ++t) {
    int ti = w + t * 8;
    int mt = ti >> 2, nt = ti & 3;
    v8f z = wmma_loop<K>(A, K, Wzt, K, mt, nt, lane);
    v8f r = wmma_loop<K>(A, K, Wrt, K, mt, nt, lane);
    v8f p = wmma_loop<DIN>(A, K, Wnt, K, mt, nt, lane);
    int n = nt * 16 + (lane & 15);
    int mbase = mt * 16 + ((lane >> 4) << 3);
    float bzv = bz[n], brv = br[n];
#pragma unroll
    for (int q = 0; q < 8; ++q) {
      int m = mbase + q;
      float hval = (m < rowsValid) ? hsrc[(rowBase + m) * 64 + n] : 0.f;
      float rg = sigmf(r[q] + brv);
      RH[m * 64 + n] = (__bf16)(rg * hval);
      z[q] = sigmf(z[q] + bzv);
    }
    zs[t] = z; ps[t] = p;
  }
  __syncthreads();
#pragma unroll
  for (int t = 0; t < 2; ++t) {
    int ti = w + t * 8;
    int mt = ti >> 2, nt = ti & 3;
    v8f qacc = wmma_loop<64>(RH, 64, Wnt + DIN, K, mt, nt, lane);
    int n = nt * 16 + (lane & 15);
    int mbase = mt * 16 + ((lane >> 4) << 3);
    float bnv = bn[n];
#pragma unroll
    for (int q = 0; q < 8; ++q) {
      int m = mbase + q;
      if (m < rowsValid) {
        long row = rowBase + m;
        float hval = hsrc[row * 64 + n];
        float c = tanhf(ps[t][q] + qacc[q] + bnv);
        float zz = zs[t][q];
        outp[row * 64 + n] = (1.f - zz) * hval + zz * c;
      }
    }
  }
}

// ---------------- utility kernels ----------------
__global__ __launch_bounds__(256) void transpose_cast_kernel(const float* __restrict__ W,
                                                             __bf16* __restrict__ Wt,
                                                             int K, int Nout, int Kpad, int Npad) {
  long i = (long)blockIdx.x * 256 + threadIdx.x;
  long total = (long)Kpad * Npad;
  if (i < total) {
    int n = (int)(i / Kpad), k = (int)(i % Kpad);
    Wt[i] = (k < K && n < Nout) ? (__bf16)W[(long)k * Nout + n] : (__bf16)0.0f;
  }
}

__global__ __launch_bounds__(256) void zero_kernel(float* p, long n) {
  long i = (long)blockIdx.x * 256 + threadIdx.x;
  if (i < n) p[i] = 0.f;
}

__global__ __launch_bounds__(256) void count_seg_kernel(const int* idxA, const int* idxB,
                                                        long rows, float* cnt) {
  long i = (long)blockIdx.x * 256 + threadIdx.x;
  if (i < rows) {
    int a = idxA[i];
    int seg = idxB ? idxB[a] : a;
    atomicAdd(&cnt[seg], 1.f);
  }
}

__global__ __launch_bounds__(256) void scatter_add_kernel(const float* __restrict__ vals,
                                                          const int* __restrict__ dstIdx,
                                                          long rows, int cols,
                                                          float* __restrict__ out) {
  long i = (long)blockIdx.x * 256 + threadIdx.x;
  if (i < rows * (long)cols) {
    long row = i / cols;
    int c = (int)(i % cols);
    atomicAdd(&out[(long)dstIdx[row] * cols + c], vals[i]);
  }
}

// LDS-pre-reduced scatter for 64-segment targets (B=64, 64 cols)
__global__ __launch_bounds__(256) void scatter_addB_kernel(const float* __restrict__ vals,
                                                           const int* __restrict__ idxA,
                                                           const int* __restrict__ idxB,
                                                           long rows, float* __restrict__ out) {
  __shared__ float acc[64 * 64];
  int tid = threadIdx.x;
  for (int i = tid; i < 4096; i += 256) acc[i] = 0.f;
  __syncthreads();
  long start = (long)blockIdx.x * 4096;
  long end = start + 4096; if (end > rows) end = rows;
  for (long i = start * 64 + tid; i < end * 64; i += 256) {
    long row = i >> 6; int c = (int)(i & 63);
    int a = idxA[row];
    int seg = idxB ? idxB[a] : a;
    atomicAdd(&acc[seg * 64 + c], vals[i]);
  }
  __syncthreads();
  for (int i = tid; i < 4096; i += 256)
    if (acc[i] != 0.f) atomicAdd(&out[i], acc[i]);
}

__global__ __launch_bounds__(256) void divide_kernel(float* sum, const float* cnt,
                                                     long rows, int cols) {
  long i = (long)blockIdx.x * 256 + threadIdx.x;
  if (i < rows * (long)cols) sum[i] /= fmaxf(cnt[i / cols], 1.f);
}

// ---------------- encoder MLP kernels ----------------
__global__ __launch_bounds__(256) void enc_edge_kernel(
    const float* __restrict__ x, const float* __restrict__ eattr, const float* __restrict__ u,
    const int* __restrict__ eidx, const int* __restrict__ batch,
    const __bf16* W1t, const float* b1, const __bf16* W2t, const float* b2,
    float* __restrict__ e1out) {
  __shared__ __bf16 A[64 * 256];
  __shared__ __bf16 H[64 * 128];
  int tid = threadIdx.x;
  long rowBase = (long)blockIdx.x * 64;
  for (int i = tid; i < 64 * 64; i += 256) {
    int row = i >> 6;
    int c4 = (i & 63) << 2;
    long e = rowBase + row; if (e >= gE) e = gE - 1;
    int s = eidx[e], d = eidx[gE + e];
    const float* sp;
    switch (c4 >> 6) {
      case 0:  sp = x + (long)s * 64; break;
      case 1:  sp = x + (long)d * 64; break;
      case 2:  sp = eattr + e * 64; break;
      default: sp = u + (long)batch[s] * 64; break;
    }
    float4 v = *(const float4*)(sp + (c4 & 63));
    __bf16* ap = A + row * 256 + c4;
    ap[0] = (__bf16)v.x; ap[1] = (__bf16)v.y; ap[2] = (__bf16)v.z; ap[3] = (__bf16)v.w;
  }
  __syncthreads();
  mlp_core64<256>(A, H, W1t, b1, W2t, b2, e1out, rowBase, gE, tid);
}

__global__ __launch_bounds__(256) void enc_node_kernel(
    const float* __restrict__ x, const float* __restrict__ meanE, const float* __restrict__ u,
    const int* __restrict__ batch,
    const __bf16* W1t, const float* b1, const __bf16* W2t, const float* b2,
    float* __restrict__ v1out) {
  __shared__ __bf16 A[64 * 192];
  __shared__ __bf16 H[64 * 128];
  int tid = threadIdx.x;
  long rowBase = (long)blockIdx.x * 64;
  for (int i = tid; i < 64 * 48; i += 256) {
    int row = i / 48;
    int c4 = (i % 48) << 2;
    long nidx = rowBase + row; if (nidx >= gN) nidx = gN - 1;
    const float* sp;
    switch (c4 >> 6) {
      case 0:  sp = x + nidx * 64; break;
      case 1:  sp = meanE + nidx * 64; break;
      default: sp = u + (long)batch[nidx] * 64; break;
    }
    float4 v = *(const float4*)(sp + (c4 & 63));
    __bf16* ap = A + row * 192 + c4;
    ap[0] = (__bf16)v.x; ap[1] = (__bf16)v.y; ap[2] = (__bf16)v.z; ap[3] = (__bf16)v.w;
  }
  __syncthreads();
  mlp_core64<192>(A, H, W1t, b1, W2t, b2, v1out, rowBase, gN, tid);
}

__global__ __launch_bounds__(256) void enc_glob_kernel(
    const float* __restrict__ m1, const float* __restrict__ m2, const float* __restrict__ u,
    const __bf16* W1t, const float* b1, const __bf16* W2t, const float* b2,
    float* __restrict__ u1out) {
  __shared__ __bf16 A[64 * 192];
  __shared__ __bf16 H[64 * 128];
  int tid = threadIdx.x;
  for (int i = tid; i < 64 * 48; i += 256) {
    int row = i / 48;
    int c4 = (i % 48) << 2;
    const float* sp;
    switch (c4 >> 6) {
      case 0:  sp = m1 + row * 64; break;
      case 1:  sp = m2 + row * 64; break;
      default: sp = u + row * 64; break;
    }
    float4 v = *(const float4*)(sp + (c4 & 63));
    __bf16* ap = A + row * 192 + c4;
    ap[0] = (__bf16)v.x; ap[1] = (__bf16)v.y; ap[2] = (__bf16)v.z; ap[3] = (__bf16)v.w;
  }
  __syncthreads();
  mlp_core64<192>(A, H, W1t, b1, W2t, b2, u1out, 0, 64, tid);
}

// ---------------- GRU kernels ----------------
__global__ __launch_bounds__(256) void rec_edge_kernel(
    const float* __restrict__ v1, const float* __restrict__ e1, const float* __restrict__ u1,
    const float* __restrict__ he, const int* __restrict__ eidx, const int* __restrict__ batch,
    const __bf16* Wzt, const __bf16* Wrt, const __bf16* Wnt,
    const float* bz, const float* br, const float* bn,
    float* __restrict__ heout) {
  __shared__ __bf16 A[64 * 320];
  __shared__ __bf16 RH[64 * 64];
  int tid = threadIdx.x;
  long rowBase = (long)blockIdx.x * 64;
  for (int i = tid; i < 64 * 80; i += 256) {
    int row = i / 80;
    int c4 = (i % 80) << 2;
    long e = rowBase + row; if (e >= gE) e = gE - 1;
    int s = eidx[e], d = eidx[gE + e];
    const float* sp;
    switch (c4 >> 6) {
      case 0:  sp = v1 + (long)s * 64; break;
      case 1:  sp = v1 + (long)d * 64; break;
      case 2:  sp = e1 + e * 64; break;
      case 3:  sp = u1 + (long)batch[s] * 64; break;
      default: sp = he + e * 64; break;
    }
    float4 v = *(const float4*)(sp + (c4 & 63));
    __bf16* ap = A + row * 320 + c4;
    ap[0] = (__bf16)v.x; ap[1] = (__bf16)v.y; ap[2] = (__bf16)v.z; ap[3] = (__bf16)v.w;
  }
  __syncthreads();
  int valid = (int)((gE - rowBase) < 64 ? (gE - rowBase) : 64);
  gru_core<320, 256>(A, RH, Wzt, Wrt, Wnt, bz, br, bn, he, heout, rowBase, valid, tid);
}

__global__ __launch_bounds__(256) void rec_node_kernel(
    const float* __restrict__ v1, const float* __restrict__ meanHe, const float* __restrict__ u1,
    const int* __restrict__ batch, const float* __restrict__ hx,
    const __bf16* Wzt, const __bf16* Wrt, const __bf16* Wnt,
    const float* bz, const float* br, const float* bn,
    float* __restrict__ hvout) {
  __shared__ __bf16 A[64 * 256];
  __shared__ __bf16 RH[64 * 64];
  int tid = threadIdx.x;
  long rowBase = (long)blockIdx.x * 64;
  for (int i = tid; i < 64 * 64; i += 256) {
    int row = i >> 6;
    int c4 = (i & 63) << 2;
    long nidx = rowBase + row; if (nidx >= gN) nidx = gN - 1;
    const float* sp;
    switch (c4 >> 6) {
      case 0:  sp = v1 + nidx * 64; break;
      case 1:  sp = meanHe + nidx * 64; break;
      case 2:  sp = u1 + (long)batch[nidx] * 64; break;
      default: sp = hx + nidx * 64; break;
    }
    float4 v = *(const float4*)(sp + (c4 & 63));
    __bf16* ap = A + row * 256 + c4;
    ap[0] = (__bf16)v.x; ap[1] = (__bf16)v.y; ap[2] = (__bf16)v.z; ap[3] = (__bf16)v.w;
  }
  __syncthreads();
  int valid = (int)((gN - rowBase) < 64 ? (gN - rowBase) : 64);
  gru_core<256, 192>(A, RH, Wzt, Wrt, Wnt, bz, br, bn, hx, hvout, rowBase, valid, tid);
}

__global__ __launch_bounds__(256) void rec_glob_kernel(
    const float* __restrict__ m1, const float* __restrict__ m2, const float* __restrict__ u1,
    const float* __restrict__ hu,
    const __bf16* Wzt, const __bf16* Wrt, const __bf16* Wnt,
    const float* bz, const float* br, const float* bn,
    float* __restrict__ huout) {
  __shared__ __bf16 A[64 * 256];
  __shared__ __bf16 RH[64 * 64];
  int tid = threadIdx.x;
  for (int i = tid; i < 64 * 64; i += 256) {
    int row = i >> 6;
    int c4 = (i & 63) << 2;
    const float* sp;
    switch (c4 >> 6) {
      case 0:  sp = m1 + row * 64; break;
      case 1:  sp = m2 + row * 64; break;
      case 2:  sp = u1 + row * 64; break;
      default: sp = hu + row * 64; break;
    }
    float4 v = *(const float4*)(sp + (c4 & 63));
    __bf16* ap = A + row * 256 + c4;
    ap[0] = (__bf16)v.x; ap[1] = (__bf16)v.y; ap[2] = (__bf16)v.z; ap[3] = (__bf16)v.w;
  }
  __syncthreads();
  gru_core<256, 192>(A, RH, Wzt, Wrt, Wnt, bz, br, bn, hu, huout, 0, 64, tid);
}

// ---------------- decoder kernels ----------------
__global__ __launch_bounds__(256) void dec_edge_kernel(
    const float* __restrict__ hv, const float* __restrict__ he, const float* __restrict__ hu,
    const int* __restrict__ eidx, const int* __restrict__ batch,
    const __bf16* W1t, const float* b1, const __bf16* W2t, const float* b2,
    const float* __restrict__ epsE,
    float* __restrict__ ozemu, float* __restrict__ ozevar, float* __restrict__ ow) {
  __shared__ __bf16 A[64 * 256];
  __shared__ __bf16 H[64 * 128];
  __shared__ float Out[64 * 16];
  int tid = threadIdx.x;
  long rowBase = (long)blockIdx.x * 64;
  for (int i = tid; i < 64 * 64; i += 256) {
    int row = i >> 6;
    int c4 = (i & 63) << 2;
    long e = rowBase + row; if (e >= gE) e = gE - 1;
    int s = eidx[e], d = eidx[gE + e];
    const float* sp;
    switch (c4 >> 6) {
      case 0:  sp = hv + (long)s * 64; break;
      case 1:  sp = hv + (long)d * 64; break;
      case 2:  sp = he + e * 64; break;
      default: sp = hu + (long)batch[s] * 64; break;
    }
    float4 v = *(const float4*)(sp + (c4 & 63));
    __bf16* ap = A + row * 256 + c4;
    ap[0] = (__bf16)v.x; ap[1] = (__bf16)v.y; ap[2] = (__bf16)v.z; ap[3] = (__bf16)v.w;
  }
  __syncthreads();
  mlp_core16<256>(A, H, W1t, b1, W2t, b2, Out, tid);
  for (int i = tid; i < 128; i += 256) {
    int row = i >> 1, j = i & 1;
    long e = rowBase + row;
    if (e < gE) {
      float mu = Out[row * 16 + j];
      float var = Out[row * 16 + 2 + j];
      float wv = sigmf(mu + expf(0.5f * var) * epsE[e * 2 + j]);
      ozemu[e * 2 + j] = mu;
      ozevar[e * 2 + j] = var;
      ow[e * 2 + j] = wv;
    }
  }
}

__global__ __launch_bounds__(256) void dec_node_kernel(
    const float* __restrict__ hv, const float* __restrict__ meanW, const float* __restrict__ hu,
    const int* __restrict__ batch,
    const __bf16* W1t, const float* b1, const __bf16* W2t, const float* b2,
    const float* __restrict__ epsN,
    float* __restrict__ oznmu, float* __restrict__ oznvar, float* __restrict__ og) {
  __shared__ __bf16 A[64 * 160];
  __shared__ __bf16 H[64 * 128];
  __shared__ float Out[64 * 16];
  int tid = threadIdx.x;
  long rowBase = (long)blockIdx.x * 64;
  for (int i = tid; i < 64 * 160; i += 256) {
    int row = i / 160;
    int c = i % 160;
    long nidx = rowBase + row; if (nidx >= gN) nidx = gN - 1;
    float val;
    if (c < 64)       val = hv[nidx * 64 + c];
    else if (c < 66)  val = meanW[nidx * 2 + (c - 64)];
    else if (c < 130) val = hu[(long)batch[nidx] * 64 + (c - 66)];
    else              val = 0.f;
    A[row * 160 + c] = (__bf16)val;
  }
  __syncthreads();
  mlp_core16<160>(A, H, W1t, b1, W2t, b2, Out, tid);
  for (int i = tid; i < 128; i += 256) {
    int row = i >> 1, j = i & 1;
    long nidx = rowBase + row;
    if (nidx < gN) {
      float mu = Out[row * 16 + j];
      float var = Out[row * 16 + 2 + j];
      float g = mu + expf(0.5f * var) * epsN[nidx * 2 + j];
      oznmu[nidx * 2 + j] = mu;
      oznvar[nidx * 2 + j] = var;
      og[nidx * 2 + j] = g;
    }
  }
}

// ---------------- host ----------------
extern "C" void kernel_launch(void* const* d_in, const int* in_sizes, int n_in,
                              void* d_out, int out_size, void* d_ws, size_t ws_size,
                              hipStream_t stream) {
  (void)in_sizes; (void)n_in; (void)out_size; (void)ws_size;
  constexpr long N = gN, E = gE;

  const float* x     = (const float*)d_in[IN_X];
  const float* eattr = (const float*)d_in[IN_EATTR];
  const float* u     = (const float*)d_in[IN_U];
  const float* hx    = (const float*)d_in[IN_HX];
  const float* hea   = (const float*)d_in[IN_HE];
  const float* huin  = (const float*)d_in[IN_HU];
  const float* epsE  = (const float*)d_in[IN_EPSE];
  const float* epsN  = (const float*)d_in[IN_EPSN];
  const int*   eidx  = (const int*)d_in[IN_EIDX];
  const int*   batch = (const int*)d_in[IN_BATCH];
  auto P = [&](int i) { return (const float*)d_in[i]; };

  // outputs (concatenated in reference return order)
  float* out = (float*)d_out;
  float* o_hv    = out;
  float* o_he    = o_hv + N * 64;
  float* o_hu    = o_he + E * 64;
  float* o_w     = o_hu + 64 * 64;
  float* o_g     = o_w + E * 2;
  float* o_zemu  = o_g + N * 2;
  float* o_zevar = o_zemu + E * 2;
  float* o_znmu  = o_zevar + E * 2;
  float* o_znvar = o_znmu + N * 2;
  (void)o_znvar;

  // workspace carve-out
  char* base = (char*)d_ws;
  size_t cur = 0;
  auto allocF = [&](size_t n) { float* p = (float*)(base + cur); cur += n * 4; cur = (cur + 255) & ~255ull; return p; };
  auto allocH = [&](size_t n) { __bf16* p = (__bf16*)(base + cur); cur += n * 2; cur = (cur + 255) & ~255ull; return p; };

  float* e1    = allocF(E * 64);
  float* v1    = allocF(N * 64);
  float* u1    = allocF(64 * 64);
  float* segN  = allocF(N * 64);
  float* segB1 = allocF(64 * 64);
  float* segB2 = allocF(64 * 64);
  float* cntN  = allocF(N);
  float* cntBn = allocF(64);
  float* cntBe = allocF(64);

  __bf16* ee_w1t = allocH(128 * 256); __bf16* ee_w2t = allocH(64 * 128);
  __bf16* en_w1t = allocH(128 * 192); __bf16* en_w2t = allocH(64 * 128);
  __bf16* eg_w1t = allocH(128 * 192); __bf16* eg_w2t = allocH(64 * 128);
  __bf16* re_wzt = allocH(64 * 320);  __bf16* re_wrt = allocH(64 * 320); __bf16* re_wnt = allocH(64 * 320);
  __bf16* rn_wzt = allocH(64 * 256);  __bf16* rn_wrt = allocH(64 * 256); __bf16* rn_wnt = allocH(64 * 256);
  __bf16* rg_wzt = allocH(64 * 256);  __bf16* rg_wrt = allocH(64 * 256); __bf16* rg_wnt = allocH(64 * 256);
  __bf16* de_w1t = allocH(128 * 256); __bf16* de_w2t = allocH(16 * 128);
  __bf16* dn_w1t = allocH(128 * 160); __bf16* dn_w2t = allocH(16 * 128);

  auto tc = [&](int pidx, __bf16* dst, int K, int Nout, int Kpad, int Npad) {
    long total = (long)Kpad * Npad;
    transpose_cast_kernel<<<(total + 255) / 256, 256, 0, stream>>>(P(pidx), dst, K, Nout, Kpad, Npad);
  };
  tc(P_EE_W1, ee_w1t, 256, 128, 256, 128); tc(P_EE_W2, ee_w2t, 128, 64, 128, 64);
  tc(P_EN_W1, en_w1t, 192, 128, 192, 128); tc(P_EN_W2, en_w2t, 128, 64, 128, 64);
  tc(P_EG_W1, eg_w1t, 192, 128, 192, 128); tc(P_EG_W2, eg_w2t, 128, 64, 128, 64);
  tc(P_RE_WZ, re_wzt, 320, 64, 320, 64);   tc(P_RE_WR, re_wrt, 320, 64, 320, 64);
  tc(P_RE_WN, re_wnt, 320, 64, 320, 64);
  tc(P_RN_WZ, rn_wzt, 256, 64, 256, 64);   tc(P_RN_WR, rn_wrt, 256, 64, 256, 64);
  tc(P_RN_WN, rn_wnt, 256, 64, 256, 64);
  tc(P_RG_WZ, rg_wzt, 256, 64, 256, 64);   tc(P_RG_WR, rg_wrt, 256, 64, 256, 64);
  tc(P_RG_WN, rg_wnt, 256, 64, 256, 64);
  tc(P_DE_W1, de_w1t, 256, 128, 256, 128); tc(P_DE_W2, de_w2t, 128, 4, 128, 16);
  tc(P_DN_W1, dn_w1t, 130, 128, 160, 128); tc(P_DN_W2, dn_w2t, 128, 4, 128, 16);

  auto zero = [&](float* p, long n) { zero_kernel<<<(n + 255) / 256, 256, 0, stream>>>(p, n); };

  // counts (computed once, reused)
  zero(cntN, N); zero(cntBn, 64); zero(cntBe, 64);
  count_seg_kernel<<<(E + 255) / 256, 256, 0, stream>>>(eidx + E, nullptr, E, cntN);
  count_seg_kernel<<<(N + 255) / 256, 256, 0, stream>>>(batch, nullptr, N, cntBn);
  count_seg_kernel<<<(E + 255) / 256, 256, 0, stream>>>(eidx, batch, E, cntBe);

  const int NB_E = (int)((E + 63) / 64);   // 5000
  const int NB_N = (int)((N + 63) / 64);   // 313

  // encoder
  enc_edge_kernel<<<NB_E, 256, 0, stream>>>(x, eattr, u, eidx, batch,
                                            ee_w1t, P(P_EE_B1), ee_w2t, P(P_EE_B2), e1);
  zero(segN, N * 64);
  scatter_add_kernel<<<(E * 64 + 255) / 256, 256, 0, stream>>>(e1, eidx + E, E, 64, segN);
  divide_kernel<<<(N * 64 + 255) / 256, 256, 0, stream>>>(segN, cntN, N, 64);
  enc_node_kernel<<<NB_N, 256, 0, stream>>>(x, segN, u, batch,
                                            en_w1t, P(P_EN_B1), en_w2t, P(P_EN_B2), v1);
  zero(segB1, 4096); zero(segB2, 4096);
  scatter_addB_kernel<<<(int)((N + 4095) / 4096), 256, 0, stream>>>(v1, batch, nullptr, N, segB1);
  scatter_addB_kernel<<<(int)((E + 4095) / 4096), 256, 0, stream>>>(e1, eidx, batch, E, segB2);
  divide_kernel<<<16, 256, 0, stream>>>(segB1, cntBn, 64, 64);
  divide_kernel<<<16, 256, 0, stream>>>(segB2, cntBe, 64, 64);
  enc_glob_kernel<<<1, 256, 0, stream>>>(segB1, segB2, u,
                                         eg_w1t, P(P_EG_B1), eg_w2t, P(P_EG_B2), u1);

  // recurrent
  rec_edge_kernel<<<NB_E, 256, 0, stream>>>(v1, e1, u1, hea, eidx, batch,
                                            re_wzt, re_wrt, re_wnt,
                                            P(P_RE_BZ), P(P_RE_BR), P(P_RE_BN), o_he);
  zero(segN, N * 64);
  scatter_add_kernel<<<(E * 64 + 255) / 256, 256, 0, stream>>>(o_he, eidx + E, E, 64, segN);
  divide_kernel<<<(N * 64 + 255) / 256, 256, 0, stream>>>(segN, cntN, N, 64);
  rec_node_kernel<<<NB_N, 256, 0, stream>>>(v1, segN, u1, batch, hx,
                                            rn_wzt, rn_wrt, rn_wnt,
                                            P(P_RN_BZ), P(P_RN_BR), P(P_RN_BN), o_hv);
  zero(segB1, 4096); zero(segB2, 4096);
  scatter_addB_kernel<<<(int)((N + 4095) / 4096), 256, 0, stream>>>(o_hv, batch, nullptr, N, segB1);
  scatter_addB_kernel<<<(int)((E + 4095) / 4096), 256, 0, stream>>>(o_he, eidx, batch, E, segB2);
  divide_kernel<<<16, 256, 0, stream>>>(segB1, cntBn, 64, 64);
  divide_kernel<<<16, 256, 0, stream>>>(segB2, cntBe, 64, 64);
  rec_glob_kernel<<<1, 256, 0, stream>>>(segB1, segB2, u1, huin,
                                         rg_wzt, rg_wrt, rg_wnt,
                                         P(P_RG_BZ), P(P_RG_BR), P(P_RG_BN), o_hu);

  // decoder (dec_glob result is discarded by the reference -> skipped)
  dec_edge_kernel<<<NB_E, 256, 0, stream>>>(o_hv, o_he, o_hu, eidx, batch,
                                            de_w1t, P(P_DE_B1), de_w2t, P(P_DE_B2),
                                            epsE, o_zemu, o_zevar, o_w);
  zero(segN, N * 2);
  scatter_add_kernel<<<(E * 2 + 255) / 256, 256, 0, stream>>>(o_w, eidx + E, E, 2, segN);
  divide_kernel<<<(N * 2 + 255) / 256, 256, 0, stream>>>(segN, cntN, N, 2);
  dec_node_kernel<<<NB_N, 256, 0, stream>>>(o_hv, segN, o_hu, batch,
                                            dn_w1t, P(P_DN_B1), dn_w2t, P(P_DN_B2),
                                            epsN, o_znmu, o_znvar, o_g);
}
